// AdaptivePathLengthCPGNN_31035433681316
// MI455X (gfx1250) — matrix operand order repaired and checked by reading, not compiled, not run on hardware
//
#include <hip/hip_runtime.h>

// ---------------------------------------------------------------------------
// AdaptivePathLengthCPGNN forward for MI455X (gfx1250), wave32 + WMMA.
// All GEMMs use v_wmma_f32_16x16x32_f16 (f32 accumulate). All B operands are
// staged as f16 *transposed* (N-major) so each B fragment is one contiguous
// 32B vector load. Attention is flash-style (no LN x LN matrices).
// ---------------------------------------------------------------------------

typedef __attribute__((ext_vector_type(16))) _Float16 v16h;
typedef __attribute__((ext_vector_type(8)))  _Float16 v8h;
typedef __attribute__((ext_vector_type(8)))  float    v8f;

#define WMMA_F16(a_, b_, c_) \
  __builtin_amdgcn_wmma_f32_16x16x32_f16(false, (a_), false, (b_), (short)0, (c_), false, false)

#define Bb   2
#define Nn   1024
#define Dd   512
#define Hh   256
#define Ll   4
#define Cc   4
#define NHh  4
#define DHd  64
#define BN   2048     // Bb*Nn
#define LN   4096     // Ll*Nn
#define BLN  8192     // Bb*LN

// ---------------------------------------------------------------------------
// Generic WMMA GEMM:  C[m][n] = act( sum_k A[m][k]*Bt[n][k] + bias[n] ) * rs[m]
// Wave tile: 16 (M) x 32 (N) -> one shared A fragment, two accumulators.
// A f32 row-major (M x K); Bt f16 transposed (N x K, row stride ldbt).
// Output: outT==0 -> f32 row-major (ldc); outT==1 -> f16 transposed
// Ct[n][m] with row stride ldc (8 contiguous halfs per lane store).
// 2-D batch (grid.y, grid.z) with independent strides (B strides in halfs).
// M multiple of 16, K multiple of 32 (true for every call here); Ncols
// arbitrary: out-of-range columns CLAMPED for loads, masked only at store.
//
// Fragment layouts used here (wave32; A and B share contiguous k order):
//   A 16x32 f16 : lane l holds row m=l&15; element e -> K = (l>>4)*16 + e
//                 => contiguous span A[m][k0 + half*16 .. +15]
//   B 32x16 f16 : lane l holds col n=l&15; element e -> K = (l>>4)*16 + e
//                 => contiguous span Bt[n][k0 + half*16 .. +15]
//   C/D 16x16 f32: VGPR r -> row m = r + (l>>4)*8, col n = l&15
// ---------------------------------------------------------------------------
__global__ void gemm_wmma_kernel(
    const float* __restrict__ A,  long syA, long szA, int lda,
    const _Float16* __restrict__ Bt, long syB, long szB, int ldbt,
    const float* __restrict__ bias, long syBias,
    const float* __restrict__ rowscale, int rsStride, long szRS,
    void* __restrict__ Cv, long syC, long szC, int ldc, int outT,
    int M, int K, int Ncols, int relu, int tnp, int tiles)
{
  int wave = threadIdx.x >> 5;
  int tile = blockIdx.x * 4 + wave;
  if (tile >= tiles) return;
  long y = blockIdx.y, z = blockIdx.z;
  A  += y * syA + z * szA;
  Bt += y * syB + z * szB;
  if (bias)     bias     += y * syBias;
  if (rowscale) rowscale += z * szRS;

  int mt = tile / tnp, ntp = tile % tnp;
  int lane = threadIdx.x & 31;
  int mr = lane & 15, half = lane >> 4;
  int gm  = mt * 16 + mr;            // A row for this lane
  int gn0 = ntp * 32 + mr;           // B/C column, first 16-tile
  int gn1 = gn0 + 16;                // second 16-tile
  bool ok0 = (gn0 < Ncols), ok1 = (gn1 < Ncols);
  int gc0 = ok0 ? gn0 : (Ncols - 1); // clamped (always in-bounds) load col
  int gc1 = ok1 ? gn1 : (Ncols - 1);

  v8f acc0 = {}, acc1 = {};
  // Pointer-based addressing: constant element offsets fold into the 24-bit
  // instruction offset; one pointer bump per operand per K-step.
  const float*    Ap  = A  + (size_t)gm * lda + half * 16;
  const _Float16* Bp0 = Bt + (size_t)gc0 * ldbt + half * 16;
  const _Float16* Bp1 = Bt + (size_t)gc1 * ldbt + half * 16;

  for (int k0 = 0; k0 < K; k0 += 32) {
    v16h af;
#pragma unroll
    for (int v = 0; v < 8; ++v) {
      int kk = v << 1;   // 0,2,4,...,14
      af[2 * v]     = (_Float16)Ap[kk];
      af[2 * v + 1] = (_Float16)Ap[kk + 1];
    }
    v16h bf0 = *(const v16h*)Bp0;
    v16h bf1 = *(const v16h*)Bp1;
    acc0 = WMMA_F16(af, bf0, acc0);
    acc1 = WMMA_F16(af, bf1, acc1);
    Ap += 32; Bp0 += 32; Bp1 += 32;
  }

  float bv0 = (bias && ok0) ? bias[gn0] : 0.f;
  float bv1 = (bias && ok1) ? bias[gn1] : 0.f;

  if (outT) {
    // f16 transposed store: Ct[n][m], lane covers rows mt*16+half*8 .. +7
    _Float16* Ct = (_Float16*)Cv + y * syC + z * szC;
    v8h h0, h1;
#pragma unroll
    for (int r = 0; r < 8; ++r) {
      int row = mt * 16 + r + half * 8;
      float rs = rowscale ? rowscale[(long)row * rsStride] : 1.f;
      float val0 = acc0[r] + bv0;
      float val1 = acc1[r] + bv1;
      if (relu) { val0 = fmaxf(val0, 0.f); val1 = fmaxf(val1, 0.f); }
      h0[r] = (_Float16)(val0 * rs);
      h1[r] = (_Float16)(val1 * rs);
    }
    int cbase = mt * 16 + half * 8;
    if (ok0) *(v8h*)(Ct + (long)gn0 * ldc + cbase) = h0;
    if (ok1) *(v8h*)(Ct + (long)gn1 * ldc + cbase) = h1;
  } else {
    float* Cm = (float*)Cv + y * syC + z * szC;
#pragma unroll
    for (int r = 0; r < 8; ++r) {
      int row = mt * 16 + r + half * 8;
      float rs = rowscale ? rowscale[(long)row * rsStride] : 1.f;
      if (ok0) {
        float val = acc0[r] + bv0;
        if (relu) val = fmaxf(val, 0.f);
        Cm[(long)row * ldc + gn0] = val * rs;
      }
      if (ok1) {
        float val = acc1[r] + bv1;
        if (relu) val = fmaxf(val, 0.f);
        Cm[(long)row * ldc + gn1] = val * rs;
      }
    }
  }
}

// ---------------------------------------------------------------------------
// Weight convert + transpose: Wt[n*K + k] = (f16) W[k*N + n]
// ---------------------------------------------------------------------------
__global__ void conv_tr_kernel(const float* __restrict__ Wsrc, _Float16* __restrict__ Wt,
                               int K, int N) {
  int i = blockIdx.x * blockDim.x + threadIdx.x;
  if (i >= K * N) return;
  int k = i / N, n = i - k * N;
  Wt[(long)n * K + k] = (_Float16)Wsrc[i];
}

// ---------------------------------------------------------------------------
// Flash attention, one wave per (b, head, 16-query tile). dh = 64.
// Q rows repeat with period N (Q broadcast over L in the reference), so we
// index qp[b*N + (ln & (N-1))] directly instead of materializing Q.
// V is staged f16-transposed: Vt[dim][key] (dim-major, row stride BLN) so
// every V fragment is one contiguous 32B load.
// ---------------------------------------------------------------------------
__global__ void flash_attn_kernel(const float* __restrict__ Qp,
                                  const float* __restrict__ Kp,
                                  const _Float16* __restrict__ Vt,
                                  float* __restrict__ Ob)
{
  __shared__ _Float16 Pl[16 * 32];
  int qt = blockIdx.x, head = blockIdx.y, b = blockIdx.z;
  int lane = threadIdx.x & 31;
  int mr = lane & 15, half = lane >> 4;
  int ln = qt * 16 + mr;

  const float* qrow = Qp + ((long)(b * Nn + (ln & (Nn - 1)))) * Hh + head * DHd + half * 16;
  v16h aq0, aq1;
#pragma unroll
  for (int v = 0; v < 8; ++v) {
    int kk = v << 1;
    aq0[2 * v]     = (_Float16)qrow[kk];
    aq0[2 * v + 1] = (_Float16)qrow[kk + 1];
    aq1[2 * v]     = (_Float16)qrow[32 + kk];
    aq1[2 * v + 1] = (_Float16)qrow[32 + kk + 1];
  }

  float mrun[8], lrun[8];
  v8f zinit = {};
  v8f oa[4];
#pragma unroll
  for (int t = 0; t < 4; ++t) oa[t] = zinit;
#pragma unroll
  for (int r = 0; r < 8; ++r) { mrun[r] = -1e30f; lrun[r] = 0.f; }

  for (int kt = 0; kt < LN / 32; ++kt) {
    // ---- S = Q K^T for 32 keys (two 16-col halves) --------------------
    v8f sfr[2];
#pragma unroll
    for (int s = 0; s < 2; ++s) {
      int key = kt * 32 + s * 16 + mr;
      const float* krow = Kp + ((long)(b * LN + key)) * Hh + head * DHd + half * 16;
      v16h b0, b1;
#pragma unroll
      for (int e = 0; e < 16; ++e) {
        b0[e] = (_Float16)krow[e];        // K^T chunk j=0 (dims 0..31)
        b1[e] = (_Float16)krow[32 + e];   // K^T chunk j=1 (dims 32..63)
      }
      v8f zacc = zinit;
      zacc = WMMA_F16(aq0, b0, zacc);
      zacc = WMMA_F16(aq1, b1, zacc);
      sfr[s] = zacc;
    }

    // ---- online softmax (rows reduce over lanes differing in bits 0..3) ---
    float p0[8], p1[8];
#pragma unroll
    for (int r = 0; r < 8; ++r) {
      float s0 = sfr[0][r] * 0.125f;   // 1/sqrt(64)
      float s1 = sfr[1][r] * 0.125f;
      float tv = fmaxf(s0, s1);
      tv = fmaxf(tv, __shfl_xor(tv, 1, 32));
      tv = fmaxf(tv, __shfl_xor(tv, 2, 32));
      tv = fmaxf(tv, __shfl_xor(tv, 4, 32));
      tv = fmaxf(tv, __shfl_xor(tv, 8, 32));
      float mnew = fmaxf(mrun[r], tv);
      float corr = __expf(mrun[r] - mnew);
      mrun[r] = mnew;
      p0[r] = __expf(s0 - mnew);
      p1[r] = __expf(s1 - mnew);
      float rs = p0[r] + p1[r];
      rs += __shfl_xor(rs, 1, 32);
      rs += __shfl_xor(rs, 2, 32);
      rs += __shfl_xor(rs, 4, 32);
      rs += __shfl_xor(rs, 8, 32);
      lrun[r] = lrun[r] * corr + rs;
#pragma unroll
      for (int t = 0; t < 4; ++t) oa[t][r] *= corr;
    }

    // ---- P (C-layout) -> LDS row-major -> A-fragment ------------------
#pragma unroll
    for (int r = 0; r < 8; ++r) {
      int m = r + half * 8;
      Pl[m * 32 + mr]      = (_Float16)p0[r];
      Pl[m * 32 + 16 + mr] = (_Float16)p1[r];
    }
    asm volatile("s_wait_dscnt 0x0" ::: "memory");
    v16h ap;
#pragma unroll
    for (int v = 0; v < 8; ++v) {
      int kk = (half << 4) + (v << 1);
      ap[2 * v]     = Pl[mr * 32 + kk];
      ap[2 * v + 1] = Pl[mr * 32 + kk + 1];
    }

    // ---- O += P @ V  (V fragments: one contiguous 32B load) -----------
#pragma unroll
    for (int t = 0; t < 4; ++t) {
      int dim = head * DHd + t * 16 + mr;
      v16h bv = *(const v16h*)(Vt + (long)dim * BLN + b * LN + kt * 32 + half * 16);
      oa[t] = WMMA_F16(ap, bv, oa[t]);
    }
  }

#pragma unroll
  for (int r = 0; r < 8; ++r) {
    float inv = 1.0f / lrun[r];
    int row = qt * 16 + r + half * 8;
    long base = ((long)(b * LN + row)) * Hh + head * DHd + mr;
    Ob[base]      = oa[0][r] * inv;
    Ob[base + 16] = oa[1][r] * inv;
    Ob[base + 32] = oa[2][r] * inv;
    Ob[base + 48] = oa[3][r] * inv;
  }
}

// ---------------------------------------------------------------------------
// Elementwise helpers
// ---------------------------------------------------------------------------
__global__ void mean_relu_kernel(const float* __restrict__ mm, float* __restrict__ outp) {
  int i = blockIdx.x * blockDim.x + threadIdx.x;
  if (i >= BN * Hh) return;
  float s = 0.f;
#pragma unroll
  for (int c = 0; c < Cc; ++c) s += mm[(long)c * BN * Hh + i];
  outp[i] = fmaxf(0.25f * s, 0.f);
}

__global__ void select_add_kernel(float* __restrict__ acc, const float* __restrict__ t,
                                  const int* __restrict__ plen, int l) {
  int i = blockIdx.x * blockDim.x + threadIdx.x;
  if (i >= BN * 256) return;
  int v = plen[i >> 8];
  v = v < 0 ? 0 : (v > 3 ? 3 : v);
  if (v == l) acc[i] += t[i];
}

__global__ void softmax4_kernel(float* __restrict__ pw) {
  int i = blockIdx.x * blockDim.x + threadIdx.x;
  if (i >= BN) return;
  float* p = pw + (long)i * 4;
  float m = fmaxf(fmaxf(p[0], p[1]), fmaxf(p[2], p[3]));
  float e0 = __expf(p[0] - m), e1 = __expf(p[1] - m);
  float e2 = __expf(p[2] - m), e3 = __expf(p[3] - m);
  float inv = 1.0f / (e0 + e1 + e2 + e3);
  p[0] = e0 * inv; p[1] = e1 * inv; p[2] = e2 * inv; p[3] = e3 * inv;
}

__global__ void sum_l_kernel(const float* __restrict__ op, float* __restrict__ osum) {
  int i = blockIdx.x * blockDim.x + threadIdx.x;
  if (i >= Bb * Nn * Hh) return;
  int b = i / (Nn * Hh);
  int r = i - b * Nn * Hh;
  long base = (long)b * Ll * Nn * Hh + r;
  float s = 0.f;
#pragma unroll
  for (int l = 0; l < Ll; ++l) s += op[base + (long)l * Nn * Hh];
  osum[i] = s;
}

__global__ void add_scaled_kernel(float* __restrict__ outb, const float* __restrict__ acc) {
  int i = blockIdx.x * blockDim.x + threadIdx.x;
  if (i >= BN * Hh) return;
  outb[i] += 0.25f * acc[i];
}

__global__ void masked_softmax_kernel(const float* __restrict__ scores,
                                      const int* __restrict__ plen,
                                      float* __restrict__ w) {
  __shared__ float sm[256];
  int tid = threadIdx.x;
  float mx = -1e30f;
  for (int i = tid; i < BN; i += 256) {
    int v = plen[i];
    if (v >= 3 && v < 5) mx = fmaxf(mx, scores[i]);
  }
  sm[tid] = mx; __syncthreads();
  for (int s = 128; s > 0; s >>= 1) { if (tid < s) sm[tid] = fmaxf(sm[tid], sm[tid + s]); __syncthreads(); }
  mx = sm[0]; __syncthreads();
  float sum = 0.f;
  for (int i = tid; i < BN; i += 256) {
    int v = plen[i];
    float e = (v >= 3 && v < 5) ? __expf(scores[i] - mx) : 0.f;
    w[i] = e; sum += e;
  }
  sm[tid] = sum; __syncthreads();
  for (int s = 128; s > 0; s >>= 1) { if (tid < s) sm[tid] += sm[tid + s]; __syncthreads(); }
  float inv = 1.0f / sm[0];
  for (int i = tid; i < BN; i += 256) w[i] *= inv;
}

__global__ void pooled_kernel(const float* __restrict__ outb, const float* __restrict__ w,
                              float* __restrict__ pooled) {
  int hh = threadIdx.x;   // 256 threads
  float s = 0.f;
  for (int i = 0; i < BN; ++i) s += outb[(long)i * Hh + hh] * w[i];
  pooled[hh] = s;
}

__global__ void final_kernel(const float* __restrict__ outb, const float* __restrict__ pooled,
                             const int* __restrict__ plen, float* __restrict__ dout) {
  int i = blockIdx.x * blockDim.x + threadIdx.x;
  if (i >= BN * Hh) return;
  int v = plen[i >> 8];
  dout[i] = (v >= 3 && v < 5) ? pooled[i & 255] : outb[i];
}

// ---------------------------------------------------------------------------
// Host side
// ---------------------------------------------------------------------------
static inline dim3 ew_grid(long total) { return dim3((unsigned)((total + 255) / 256)); }

static void gemm(hipStream_t st,
                 const float* A, long syA, long szA, int lda,
                 const _Float16* Bt, long syB, long szB, int ldbt,
                 const float* bias, long syBias,
                 const float* rowscale, int rsStride, long szRS,
                 void* Cv, long syC, long szC, int ldc, int outT,
                 int M, int K, int Ncols, int relu, int ny, int nz)
{
  int tm = M / 16, tnp = (Ncols + 31) / 32;
  int tiles = tm * tnp;
  dim3 grid((tiles + 3) / 4, ny, nz);
  hipLaunchKernelGGL(gemm_wmma_kernel, grid, dim3(128), 0, st,
                     A, syA, szA, lda, Bt, syB, szB, ldbt,
                     bias, syBias, rowscale, rsStride, szRS,
                     Cv, syC, szC, ldc, outT, M, K, Ncols, relu, tnp, tiles);
}

static void ctr(hipStream_t st, const float* Wsrc, _Float16* Wt, int K, int N) {
  hipLaunchKernelGGL(conv_tr_kernel, ew_grid((long)K * N), dim3(256), 0, st, Wsrc, Wt, K, N);
}

extern "C" void kernel_launch(void* const* d_in, const int* in_sizes, int n_in,
                              void* d_out, int out_size, void* d_ws, size_t ws_size,
                              hipStream_t stream)
{
  (void)in_sizes; (void)n_in; (void)out_size; (void)ws_size;
  int pi = 0;
  auto nxt = [&]() { return (const float*)d_in[pi++]; };

  const float* x    = nxt();
  const float* adj  = nxt();
  const int*   plen = (const int*)d_in[pi++];
  const float* neW = nxt();  const float* neB = nxt();
  const float* msgW = nxt(); const float* msgB = nxt();

  static const int mlp_nl[Ll] = {3, 3, 4, 4};
  static const int mlp_dims[Ll][5] = {
    {256, 256, 192, 256, 0}, {256, 256, 192, 256, 0},
    {256, 512, 384, 256, 256}, {256, 512, 384, 256, 256}};
  const float* mlpW[Cc][Ll][4]; const float* mlpB[Cc][Ll][4];
  for (int c = 0; c < Cc; ++c)
    for (int l = 0; l < Ll; ++l)
      for (int j = 0; j < mlp_nl[l]; ++j) { mlpW[c][l][j] = nxt(); mlpB[c][l][j] = nxt(); }

  const float* qW = nxt();  const float* qB = nxt();
  const float* kW[Ll]; const float* kB[Ll];
  for (int l = 0; l < Ll; ++l) { kW[l] = nxt(); kB[l] = nxt(); }
  const float* vW[Ll]; const float* vB[Ll];
  for (int l = 0; l < Ll; ++l) { vW[l] = nxt(); vB[l] = nxt(); }
  const float* mqW = nxt(); const float* mqB = nxt();
  const float* mkW = nxt(); const float* mkB = nxt();
  const float* mvW = nxt(); const float* mvB = nxt();
  const float* moW = nxt(); const float* moB = nxt();
  const float* i1W = nxt(); const float* i1B = nxt();
  const float* i2W = nxt(); const float* i2B = nxt();
  const float* oW  = nxt(); const float* oB  = nxt();
  const float* p1W = nxt(); const float* p1B = nxt();
  const float* p2W = nxt(); const float* p2B = nxt();

  // ---- f32 workspace (lifetime-based reuse) ------------------------------
  float* W = (float*)d_ws;
  size_t off = 0;
  auto alloc = [&](size_t n) { float* p = W + off; off += n; return p; };
  float* h    = alloc((size_t)BN * Hh);
  float* pe[Ll];
  for (int l = 0; l < Ll; ++l) pe[l] = alloc((size_t)BN * Hh);
  float* R1   = alloc((size_t)Cc * BN * Hh);  // mbuf16 -> Kbuf -> obuf
  float* R2   = alloc((size_t)Cc * BN * Hh);  // mm -> t0/t1 -> Vbuf -> oproj
  float* accb = alloc((size_t)BN * Hh);
  float* qbuf = alloc((size_t)BN * Hh);
  float* impt = alloc((size_t)BN * 128);
  float* pw   = alloc((size_t)BN * 4);
  float* Qp   = alloc((size_t)BN * Hh);
  float* Kp   = alloc((size_t)Bb * LN * Hh);
  float* osum = alloc((size_t)BN * Hh);
  float* outb = alloc((size_t)BN * Hh);
  float* poolt = alloc((size_t)BN * 128);
  float* scores = alloc((size_t)BN);
  float* wv     = alloc((size_t)BN);
  float* pooled = alloc((size_t)Hh);
  _Float16* Vt = (_Float16*)alloc((size_t)Hh * BLN / 2);   // f16 V, dim-major

  // ---- f16 weight arena (transposed N-major copies) ----------------------
  _Float16* HA = (_Float16*)alloc(8 * 1024 * 1024 / 2);    // 8M halfs
  size_t hoff = 0;
  auto allocH = [&](size_t n) { _Float16* p = HA + hoff; hoff += n; return p; };
  _Float16* neWt  = allocH((size_t)Dd * Hh);
  _Float16* msgWt = allocH((size_t)Ll * Cc * Hh * Hh);
  _Float16* mlpWt[Cc][Ll][4];
  for (int c = 0; c < Cc; ++c)
    for (int l = 0; l < Ll; ++l)
      for (int j = 0; j < mlp_nl[l]; ++j)
        mlpWt[c][l][j] = allocH((size_t)mlp_dims[l][j] * mlp_dims[l][j + 1]);
  _Float16* qWt = allocH((size_t)Hh * Hh);
  _Float16* kWt[Ll]; _Float16* vWt[Ll];
  for (int l = 0; l < Ll; ++l) { kWt[l] = allocH((size_t)Hh * Hh); vWt[l] = allocH((size_t)Hh * Hh); }
  _Float16* mqWt = allocH((size_t)Hh * Hh);
  _Float16* mkWt = allocH((size_t)Hh * Hh);
  _Float16* mvWt = allocH((size_t)Hh * Hh);
  _Float16* moWt = allocH((size_t)Hh * Hh);
  _Float16* i1Wt = allocH((size_t)Hh * 128);
  _Float16* i2Wt = allocH((size_t)128 * 4);
  _Float16* oWt  = allocH((size_t)Hh * Hh);
  _Float16* p1Wt = allocH((size_t)Hh * 128);
  _Float16* p2Wt = allocH((size_t)128 * 1);

  // ---- convert all weights once per launch (deterministic) ---------------
  ctr(stream, neW, neWt, Dd, Hh);
  for (int i = 0; i < Ll * Cc; ++i)
    ctr(stream, msgW + (size_t)i * Hh * Hh, msgWt + (size_t)i * Hh * Hh, Hh, Hh);
  for (int c = 0; c < Cc; ++c)
    for (int l = 0; l < Ll; ++l)
      for (int j = 0; j < mlp_nl[l]; ++j)
        ctr(stream, mlpW[c][l][j], mlpWt[c][l][j], mlp_dims[l][j], mlp_dims[l][j + 1]);
  ctr(stream, qW, qWt, Hh, Hh);
  for (int l = 0; l < Ll; ++l) { ctr(stream, kW[l], kWt[l], Hh, Hh); ctr(stream, vW[l], vWt[l], Hh, Hh); }
  ctr(stream, mqW, mqWt, Hh, Hh);  ctr(stream, mkW, mkWt, Hh, Hh);
  ctr(stream, mvW, mvWt, Hh, Hh);  ctr(stream, moW, moWt, Hh, Hh);
  ctr(stream, i1W, i1Wt, Hh, 128); ctr(stream, i2W, i2Wt, 128, 4);
  ctr(stream, oW, oWt, Hh, Hh);
  ctr(stream, p1W, p1Wt, Hh, 128); ctr(stream, p2W, p2Wt, 128, 1);

  // ---- Stage 0: node embedding  h = x @ neW + neB ------------------------
  gemm(stream, x, 0, 0, Dd, neWt, 0, 0, Dd, neB, 0, nullptr, 0, 0,
       h, 0, 0, Hh, 0, BN, Dd, Hh, 0, 1, 1);

  // ---- Stage 1: message passing ------------------------------------------
  _Float16* mbuf16 = (_Float16*)R1;   // f16 transposed message tensor
  float* mmb = R2;
  const float* cur = h;
  for (int l = 0; l < Ll; ++l) {
    // m16[c][h][b*N+node] = (cur @ msg_W[l][c] + msg_b[l][c])  (f16 transposed out)
    gemm(stream, cur, 0, 0, Hh,
         msgWt + (size_t)l * Cc * Hh * Hh, (long)Hh * Hh, 0, Hh,
         msgB + (size_t)l * Cc * Hh, Hh, nullptr, 0, 0,
         mbuf16, (long)Hh * BN, 0, BN, 1,
         BN, Hh, Hh, 0, Cc, 1);
    // mm[c,b] = adj[b] @ m[c,b]   (B = f16 transposed, batched c (y), b (z))
    gemm(stream, adj, 0, (long)Nn * Nn, Nn,
         mbuf16, (long)Hh * BN, (long)Nn, BN,
         nullptr, 0, nullptr, 0, 0,
         mmb, (long)Bb * Nn * Hh, (long)Nn * Hh, Hh, 0,
         Nn, Nn, Hh, 0, Cc, Bb);
    hipLaunchKernelGGL(mean_relu_kernel, ew_grid((long)BN * Hh), dim3(256), 0, stream, mmb, pe[l]);
    cur = pe[l];
  }

  // ---- Stage 2: adaptive per-bucket MLPs ----------------------------------
  hipMemsetAsync(accb, 0, (size_t)BN * Hh * sizeof(float), stream);
  float* t0 = R2;
  float* t1 = R2 + (size_t)BN * 512;
  for (int c = 0; c < Cc; ++c) {
    for (int l = 0; l < Ll; ++l) {
      const float* z = h;
      int nl = mlp_nl[l];
      for (int j = 0; j < nl; ++j) {
        int din = mlp_dims[l][j], dout_ = mlp_dims[l][j + 1];
        float* dst = (j & 1) ? t1 : t0;
        gemm(stream, z, 0, 0, din, mlpWt[c][l][j], 0, 0, din,
             mlpB[c][l][j], 0, nullptr, 0, 0,
             dst, 0, 0, dout_, 0, BN, din, dout_, (j < nl - 1) ? 1 : 0, 1, 1);
        z = dst;
      }
      hipLaunchKernelGGL(select_add_kernel, ew_grid((long)BN * 256), dim3(256), 0, stream,
                         accb, (float*)z, plen, l);
    }
  }

  // ---- Stage 3: attention branch -----------------------------------------
  gemm(stream, h, 0, 0, Hh, qWt, 0, 0, Hh, qB, 0, nullptr, 0, 0,
       qbuf, 0, 0, Hh, 0, BN, Hh, Hh, 0, 1, 1);
  gemm(stream, h, 0, 0, Hh, i1Wt, 0, 0, Hh, i1B, 0, nullptr, 0, 0,
       impt, 0, 0, 128, 0, BN, Hh, 128, 1, 1, 1);
  gemm(stream, impt, 0, 0, 128, i2Wt, 0, 0, 128, i2B, 0, nullptr, 0, 0,
       pw, 0, 0, 4, 0, BN, 128, 4, 0, 1, 1);
  hipLaunchKernelGGL(softmax4_kernel, ew_grid(BN), dim3(256), 0, stream, pw);

  float* Kbuf = R1;   // mbuf16 dead
  float* Vbuf = R2;   // t0/t1 dead
  for (int l = 0; l < Ll; ++l) {
    // K_l[b] = (pe[l][b] @ kW + kB) * pw[:, :, l]
    gemm(stream, pe[l], 0, (long)Nn * Hh, Hh, kWt[l], 0, 0, Hh, kB[l], 0,
         pw + l, 4, (long)Nn * 4,
         Kbuf + (size_t)l * Nn * Hh, 0, (long)Ll * Nn * Hh, Hh, 0,
         Nn, Hh, Hh, 0, 1, Bb);
    gemm(stream, pe[l], 0, (long)Nn * Hh, Hh, vWt[l], 0, 0, Hh, vB[l], 0,
         pw + l, 4, (long)Nn * 4,
         Vbuf + (size_t)l * Nn * Hh, 0, (long)Ll * Nn * Hh, Hh, 0,
         Nn, Hh, Hh, 0, 1, Bb);
  }
  gemm(stream, qbuf, 0, 0, Hh, mqWt, 0, 0, Hh, mqB, 0, nullptr, 0, 0,
       Qp, 0, 0, Hh, 0, BN, Hh, Hh, 0, 1, 1);
  gemm(stream, Kbuf, 0, 0, Hh, mkWt, 0, 0, Hh, mkB, 0, nullptr, 0, 0,
       Kp, 0, 0, Hh, 0, BLN, Hh, Hh, 0, 1, 1);
  // V projection written f16 transposed: Vt[dim][key], ld = BLN
  gemm(stream, Vbuf, 0, 0, Hh, mvWt, 0, 0, Hh, mvB, 0, nullptr, 0, 0,
       Vt, 0, 0, BLN, 1, BLN, Hh, Hh, 0, 1, 1);

  float* obuf = R1;   // Kbuf dead after Kp
  hipLaunchKernelGGL(flash_attn_kernel, dim3(LN / 16, NHh, Bb), dim3(32), 0, stream,
                     Qp, Kp, Vt, obuf);

  float* oproj = R2;  // Vbuf dead after Vt
  gemm(stream, obuf, 0, 0, Hh, moWt, 0, 0, Hh, moB, 0, nullptr, 0, 0,
       oproj, 0, 0, Hh, 0, BLN, Hh, Hh, 0, 1, 1);
  hipLaunchKernelGGL(sum_l_kernel, ew_grid((long)BN * Hh), dim3(256), 0, stream, oproj, osum);

  gemm(stream, osum, 0, 0, Hh, oWt, 0, 0, Hh, oB, 0, nullptr, 0, 0,
       outb, 0, 0, Hh, 0, BN, Hh, Hh, 0, 1, 1);
  hipLaunchKernelGGL(add_scaled_kernel, ew_grid((long)BN * Hh), dim3(256), 0, stream, outb, accb);

  // ---- Stage 4: masked softmax pooling ------------------------------------
  gemm(stream, outb, 0, 0, Hh, p1Wt, 0, 0, Hh, p1B, 0, nullptr, 0, 0,
       poolt, 0, 0, 128, 0, BN, Hh, 128, 1, 1, 1);
  gemm(stream, poolt, 0, 0, 128, p2Wt, 0, 0, 128, p2B, 0, nullptr, 0, 0,
       scores, 0, 0, 1, 0, BN, 128, 1, 0, 1, 1);
  hipLaunchKernelGGL(masked_softmax_kernel, dim3(1), dim3(256), 0, stream, scores, plen, wv);
  hipLaunchKernelGGL(pooled_kernel, dim3(1), dim3(256), 0, stream, outb, wv, pooled);
  hipLaunchKernelGGL(final_kernel, ew_grid((long)BN * Hh), dim3(256), 0, stream,
                     outb, pooled, plen, (float*)d_out);
}
